// ReverseLstmLayer_80187039416469
// MI455X (gfx1250) — matrix operand order repaired and checked by reading, not compiled, not run on hardware
//
#include <hip/hip_runtime.h>
#include <hip/hip_bf16.h>
#include <math.h>

// ---------------------------------------------------------------------------
// Reverse-direction field-gated LSTM for MI455X (gfx1250), bf16 WMMA path.
//   Phase 1a: pregate[B*T,4H] = seq @ Wx + b              (bf16 WMMA, f32 acc)
//   Phase 1b: fadd[B*T,H]     = sigmoid(fp@Wf+bf |l) * tanh(fp@Wf+bf |z)
//   Phase 2 : 256 sequential steps: rg = h @ Wh (bf16 WMMA) + fused cell.
// 128x128 macro-tiles, 8 WMMAs/wave/K-step, b128 LDS fragment loads,
// register-prefetched K pipeline.
// ---------------------------------------------------------------------------

typedef __bf16 bf16_t;
typedef __attribute__((ext_vector_type(16))) __bf16    v16bf;
typedef __attribute__((ext_vector_type(8)))  float     v8f;
typedef __attribute__((ext_vector_type(4)))  float     f32x4;
typedef __attribute__((ext_vector_type(4)))  uint32_t  u32x4;
typedef __attribute__((ext_vector_type(2)))  uint32_t  u32x2;

constexpr int kB  = 64;
constexpr int kT  = 256;
constexpr int kD  = 1024;
constexpr int kH  = 1024;
constexpr int kBT = kB * kT;     // 16384
constexpr int kG4 = 4 * kH;      // 4096
constexpr int kG2 = 2 * kH;      // 2048

__device__ __forceinline__ float sigmoidf(float x) {
  return 1.0f / (1.0f + __expf(-x));
}

__device__ __forceinline__ uint32_t pack2bf(float a, float b) {
  union { bf16_t h[2]; uint32_t u; } p;
  p.h[0] = (bf16_t)a; p.h[1] = (bf16_t)b;
  return p.u;
}

// --- LDS layouts ------------------------------------------------------------
// A tile: row-major bf16, row stride 20 dwords (40 bf16): 16B-aligned rows,
//         20 mod 64 banks -> conflict-free across 16 lanes.
// B tile: fragment-order u32 K-pairs: ldsB[n*20 + kh*8 + v] holds
//         {K=kh*16+2v (lo), +1 (hi)} for column n.
// Both give fragments as two contiguous 16B chunks -> ds_load_b128.

__device__ __forceinline__ v16bf frag_a(const uint32_t* ldsA, int mrow0, int lane) {
  const int l16 = lane & 15, kh = (lane >> 4) & 1;
  const uint32_t* base = ldsA + (mrow0 + l16) * 20 + kh * 8;
  union { u32x4 q[2]; v16bf v; } r;
  r.q[0] = *(const u32x4*)(base);       // K = kh*16 + 0..7
  r.q[1] = *(const u32x4*)(base + 4);   // K = kh*16 + 8..15
  return r.v;
}

__device__ __forceinline__ v16bf frag_b(const uint32_t* ldsB, int ncol0, int lane) {
  const int l16 = lane & 15, kh = (lane >> 4) & 1;
  const uint32_t* base = ldsB + (ncol0 + l16) * 20 + kh * 8;
  union { u32x4 q[2]; v16bf v; } r;
  r.q[0] = *(const u32x4*)(base);       // K = kh*16 + 0..7
  r.q[1] = *(const u32x4*)(base + 4);   // K = kh*16 + 8..15
  return r.v;
}

#define WMMA_BF16(Afrag, Bfrag, Cacc) \
  __builtin_amdgcn_wmma_f32_16x16x32_bf16(false, (Afrag), false, (Bfrag), (short)0, (Cacc), false, false)

// ---------------------------------------------------------------------------
// Phase 1a: C[M,N] = bf16(A[M,K]) x bf16(W[K,N]) + bias.  Block = 128x128,
// 8 waves in 2x4 grid, each wave 4(M) x 2(N) tiles = 8 WMMAs / K-step.
// ---------------------------------------------------------------------------
__global__ __launch_bounds__(256)
void gemm_xw_kernel(const float* __restrict__ A, const float* __restrict__ W,
                    const float* __restrict__ bias, float* __restrict__ C,
                    int K, int N) {
  __shared__ alignas(16) uint32_t smem[2560 + 2560];  // A:128*20, B:128*20
  uint32_t* ldsA = smem;
  uint32_t* ldsB = smem + 2560;

  const int tid  = threadIdx.x;
  const int lane = tid & 31;
  const int wave = tid >> 5;
  const int wr   = wave >> 2;        // 0..1
  const int wc   = wave & 3;         // 0..3
  const int n0   = blockIdx.x * 128;
  const int m0   = blockIdx.y * 128;

  v8f acc[4][2] = {};
  f32x4 pa[4];                       // A prefetch: 128x32 / 256thr = 4 float4
  f32x4 pb[2][2];                    // B prefetch: 2 slots x {row k, row k+1}

  auto fetch = [&](int kk) {
#pragma unroll
    for (int r = 0; r < 4; ++r) {
      int idx = tid + r * 256;       // 1024 float4 slots
      int row = idx >> 3, c4 = idx & 7;
      pa[r] = *(const f32x4*)(A + (size_t)(m0 + row) * K + kk + c4 * 4);
    }
#pragma unroll
    for (int s = 0; s < 2; ++s) {    // 16 kp x 32 col-quads = 512 slots
      int slot = tid + s * 256;
      int kp = slot >> 5, ng = slot & 31;
      const float* w0 = W + (size_t)(kk + 2 * kp) * N + n0 + ng * 4;
      pb[s][0] = *(const f32x4*)(w0);
      pb[s][1] = *(const f32x4*)(w0 + N);
    }
  };
  auto stage = [&]() {
#pragma unroll
    for (int r = 0; r < 4; ++r) {
      int idx = tid + r * 256;
      int row = idx >> 3, c4 = idx & 7;
      u32x2 d; d.x = pack2bf(pa[r].x, pa[r].y); d.y = pack2bf(pa[r].z, pa[r].w);
      *(u32x2*)(ldsA + row * 20 + c4 * 2) = d;
    }
#pragma unroll
    for (int s = 0; s < 2; ++s) {
      int slot = tid + s * 256;
      int kp = slot >> 5, ng = slot & 31;
      int kh = kp >> 3, v = kp & 7;
#pragma unroll
      for (int j = 0; j < 4; ++j)
        ldsB[(ng * 4 + j) * 20 + kh * 8 + v] = pack2bf(pb[s][0][j], pb[s][1][j]);
    }
  };

  fetch(0);
  for (int kk = 0; kk < K; kk += 32) {
    __syncthreads();
    stage();
    __syncthreads();
    if (kk + 32 < K) fetch(kk + 32);          // overlap globals with WMMAs
    v16bf af[4], bfr[2];
#pragma unroll
    for (int im = 0; im < 4; ++im) af[im] = frag_a(ldsA, (wr * 4 + im) * 16, lane);
#pragma unroll
    for (int in = 0; in < 2; ++in) bfr[in] = frag_b(ldsB, (wc * 2 + in) * 16, lane);
#pragma unroll
    for (int im = 0; im < 4; ++im)
#pragma unroll
      for (int in = 0; in < 2; ++in)
        acc[im][in] = WMMA_BF16(af[im], bfr[in], acc[im][in]);
  }

  const int l16 = lane & 15, kh = (lane >> 4) & 1;
#pragma unroll
  for (int im = 0; im < 4; ++im)
#pragma unroll
    for (int in = 0; in < 2; ++in)
#pragma unroll
      for (int r = 0; r < 8; ++r) {
        int m = m0 + (wr * 4 + im) * 16 + r + 8 * kh;
        int c = n0 + (wc * 2 + in) * 16 + l16;
        C[(size_t)m * N + c] = acc[im][in][r] + bias[c];
      }
}

// ---------------------------------------------------------------------------
// Phase 1b: field GEMM, fused epilogue. Block = 128 rows x 64 cols of BOTH
// halves (l: n0.., z: 1024+n0..). Wave = 4(M) x 1(N) x 2(halves) = 8 WMMAs.
// ---------------------------------------------------------------------------
__global__ __launch_bounds__(256)
void gemm_field_kernel(const float* __restrict__ A, const float* __restrict__ Wf,
                       const float* __restrict__ bfv, float* __restrict__ fadd,
                       int K) {
  __shared__ alignas(16) uint32_t smem[2560 + 1280 + 1280];
  uint32_t* ldsA = smem;             // 128 x 20
  uint32_t* ldsL = smem + 2560;      // 64 x 20
  uint32_t* ldsZ = smem + 3840;      // 64 x 20

  const int tid  = threadIdx.x;
  const int lane = tid & 31;
  const int wave = tid >> 5;
  const int wr   = wave >> 2;
  const int wc   = wave & 3;         // ni = wc (one 16-col group)
  const int n0   = blockIdx.x * 64;  // within H
  const int m0   = blockIdx.y * 128;

  v8f accL[4] = {}, accZ[4] = {};
  f32x4 pa[4];
  f32x4 plo[2], pzo[2];              // one slot per thread per half: rows k,k+1

  auto fetch = [&](int kk) {
#pragma unroll
    for (int r = 0; r < 4; ++r) {
      int idx = tid + r * 256;
      int row = idx >> 3, c4 = idx & 7;
      pa[r] = *(const f32x4*)(A + (size_t)(m0 + row) * K + kk + c4 * 4);
    }
    int kp = tid >> 4, ng = tid & 15;          // 16 kp x 16 col-quads = 256
    const float* w0 = Wf + (size_t)(kk + 2 * kp) * kG2 + n0 + ng * 4;
    plo[0] = *(const f32x4*)(w0);
    plo[1] = *(const f32x4*)(w0 + kG2);
    pzo[0] = *(const f32x4*)(w0 + kH);
    pzo[1] = *(const f32x4*)(w0 + kG2 + kH);
  };
  auto stage = [&]() {
#pragma unroll
    for (int r = 0; r < 4; ++r) {
      int idx = tid + r * 256;
      int row = idx >> 3, c4 = idx & 7;
      u32x2 d; d.x = pack2bf(pa[r].x, pa[r].y); d.y = pack2bf(pa[r].z, pa[r].w);
      *(u32x2*)(ldsA + row * 20 + c4 * 2) = d;
    }
    int kp = tid >> 4, ng = tid & 15;
    int kh = kp >> 3, v = kp & 7;
#pragma unroll
    for (int j = 0; j < 4; ++j) {
      ldsL[(ng * 4 + j) * 20 + kh * 8 + v] = pack2bf(plo[0][j], plo[1][j]);
      ldsZ[(ng * 4 + j) * 20 + kh * 8 + v] = pack2bf(pzo[0][j], pzo[1][j]);
    }
  };

  fetch(0);
  for (int kk = 0; kk < K; kk += 32) {
    __syncthreads();
    stage();
    __syncthreads();
    if (kk + 32 < K) fetch(kk + 32);
    v16bf bl = frag_b(ldsL, wc * 16, lane);
    v16bf bz = frag_b(ldsZ, wc * 16, lane);
#pragma unroll
    for (int im = 0; im < 4; ++im) {
      v16bf a = frag_a(ldsA, (wr * 4 + im) * 16, lane);
      accL[im] = WMMA_BF16(a, bl, accL[im]);
      accZ[im] = WMMA_BF16(a, bz, accZ[im]);
    }
  }

  const int l16 = lane & 15, kh = (lane >> 4) & 1;
#pragma unroll
  for (int im = 0; im < 4; ++im)
#pragma unroll
    for (int r = 0; r < 8; ++r) {
      int m = m0 + (wr * 4 + im) * 16 + r + 8 * kh;
      int c = n0 + wc * 16 + l16;
      float l = accL[im][r] + bfv[c];
      float z = accZ[im][r] + bfv[kH + c];
      fadd[(size_t)m * kH + c] = sigmoidf(l) * tanhf(z);
    }
}

// ---------------------------------------------------------------------------
// Phase 2: one recurrent step (launched 256x, stream-ordered).
// Block b owns h-columns [16b,16b+16): rgate[64 batch][4 gates x 16] via bf16
// WMMA over K=1024, then fused LSTM cell update. h ping-pongs in bf16.
// ---------------------------------------------------------------------------
__global__ __launch_bounds__(256)
void lstm_step_kernel(const bf16_t* __restrict__ hsrc, bf16_t* __restrict__ hdst,
                      const bf16_t* __restrict__ Whbf,
                      const float* __restrict__ pregate,
                      const float* __restrict__ fadd,
                      float* __restrict__ cstate,
                      float* __restrict__ outs, int t) {
  __shared__ alignas(16) uint32_t smem[4096];  // GEMM: A 1280 + B 1280; gates 4096 f32
  uint32_t* ldsA = smem;             // 64 rows x 20 dw
  uint32_t* ldsB = smem + 1280;      // 64 cols x 20 dw
  float*    gsh  = (float*)smem;     // reused after GEMM

  const int tid  = threadIdx.x;
  const int lane = tid & 31;
  const int wave = tid >> 5;
  const int hc0  = blockIdx.x * 16;
  const int mi   = (wave * 2) >> 2;  // batch tile 0..3
  const int ni0  = (wave * 2) & 3;   // gate pair start (0 or 2)

  v8f acc0 = {}, acc1 = {};
  u32x2 pa[2];                       // h: 64x32 bf16 = 512 quads / 256thr = 2
  u32x2 pb0, pb1;                    // Wh: one (kp, col-quad) slot per thread

  auto fetch = [&](int kk) {
#pragma unroll
    for (int r = 0; r < 2; ++r) {
      int q = tid + r * 256;
      int row = q >> 3, c4 = q & 7;
      pa[r] = *(const u32x2*)(hsrc + row * kH + kk + c4 * 4);
    }
    int kp = tid >> 4, cq = tid & 15;            // 16 kp x 16 col-quads
    int g = cq >> 2, c4 = (cq & 3) * 4;
    const bf16_t* w0 = Whbf + (size_t)(kk + 2 * kp) * kG4 + g * kH + hc0 + c4;
    pb0 = *(const u32x2*)(w0);
    pb1 = *(const u32x2*)(w0 + kG4);
  };
  auto stage = [&]() {
#pragma unroll
    for (int r = 0; r < 2; ++r) {
      int q = tid + r * 256;
      int row = q >> 3, c4 = q & 7;
      *(u32x2*)(ldsA + row * 20 + c4 * 2) = pa[r];
    }
    int kp = tid >> 4, cq = tid & 15;
    int kh = kp >> 3, v = kp & 7;
    union { u32x2 d; bf16_t h[4]; } r0, r1;
    r0.d = pb0; r1.d = pb1;
#pragma unroll
    for (int j = 0; j < 4; ++j) {
      union { bf16_t h[2]; uint32_t u; } p;
      p.h[0] = r0.h[j]; p.h[1] = r1.h[j];
      ldsB[(cq * 4 + j) * 20 + kh * 8 + v] = p.u;
    }
  };

  fetch(0);
  for (int kk = 0; kk < kH; kk += 32) {
    __syncthreads();
    stage();
    __syncthreads();
    if (kk + 32 < kH) fetch(kk + 32);
    v16bf a  = frag_a(ldsA, mi * 16, lane);
    v16bf b0 = frag_b(ldsB, ni0 * 16, lane);
    v16bf b1 = frag_b(ldsB, (ni0 + 1) * 16, lane);
    acc0 = WMMA_BF16(a, b0, acc0);
    acc1 = WMMA_BF16(a, b1, acc1);
  }

  __syncthreads();                   // GEMM LDS dead; reuse for gates
  const int l16 = lane & 15, kh = (lane >> 4) & 1;
#pragma unroll
  for (int r = 0; r < 8; ++r) {
    int m = mi * 16 + r + 8 * kh;
    gsh[m * 64 + ni0 * 16 + l16]       = acc0[r];
    gsh[m * 64 + (ni0 + 1) * 16 + l16] = acc1[r];
  }
  __syncthreads();

#pragma unroll
  for (int r = 0; r < 4; ++r) {      // 64 batch x 16 cols = 1024 items
    int idx = tid + r * 256;
    int bm = idx >> 4, c = idx & 15;
    int hcol = hc0 + c;
    size_t prow = ((size_t)bm * kT + t) * kG4;
    float gi = gsh[bm * 64 +  0 + c] + pregate[prow          + hcol];
    float gf = gsh[bm * 64 + 16 + c] + pregate[prow + 1 * kH + hcol];
    float gg = gsh[bm * 64 + 32 + c] + pregate[prow + 2 * kH + hcol];
    float go = gsh[bm * 64 + 48 + c] + pregate[prow + 3 * kH + hcol];
    float cold = cstate[bm * kH + hcol];
    float cnew = sigmoidf(gf) * cold + sigmoidf(gi) * tanhf(gg)
               + fadd[((size_t)bm * kT + t) * kH + hcol];
    float hnew = sigmoidf(go) * tanhf(cnew);
    cstate[bm * kH + hcol] = cnew;
    hdst[bm * kH + hcol]   = (bf16_t)hnew;
    outs[((size_t)bm * kT + t) * kH + hcol] = hnew;
  }
}

// ---------------------------------------------------------------------------
// Small helper kernels.
// ---------------------------------------------------------------------------
__global__ __launch_bounds__(256)
void convert_wh_kernel(const float* __restrict__ Wh, bf16_t* __restrict__ Whbf) {
  int i0 = (blockIdx.x * 256 + threadIdx.x) * 4;
  f32x4 f = *(const f32x4*)(Wh + i0);
  u32x2 d; d.x = pack2bf(f.x, f.y); d.y = pack2bf(f.z, f.w);
  *(u32x2*)(Whbf + i0) = d;
}

__global__ __launch_bounds__(256)
void init_state_kernel(const float* __restrict__ h0, const float* __restrict__ c0,
                       bf16_t* __restrict__ hbf, float* __restrict__ cstate) {
  int i = blockIdx.x * 256 + threadIdx.x;      // 0 .. 64*1024-1
  hbf[i]    = (bf16_t)h0[i];
  cstate[i] = c0[i];
}

__global__ __launch_bounds__(256)
void finalize_kernel(const float* __restrict__ outs, const float* __restrict__ cstate,
                     float* __restrict__ dst_h, float* __restrict__ dst_c) {
  int i = blockIdx.x * 256 + threadIdx.x;      // 0 .. 64*1024-1
  int b = i >> 10, h = i & 1023;
  dst_h[i] = outs[((size_t)b * kT) * kH + h];  // h_final = outputs[:, 0, :]
  dst_c[i] = cstate[i];
}

// ---------------------------------------------------------------------------
extern "C" void kernel_launch(void* const* d_in, const int* in_sizes, int n_in,
                              void* d_out, int out_size, void* d_ws, size_t ws_size,
                              hipStream_t stream) {
  (void)in_sizes; (void)n_in; (void)out_size; (void)ws_size;

  const float* seq = (const float*)d_in[0];   // [B,T,D]
  const float* fld = (const float*)d_in[1];   // [B,T,F]
  const float* h0  = (const float*)d_in[2];   // [B,H]
  const float* c0  = (const float*)d_in[3];   // [B,H]
  const float* Wx  = (const float*)d_in[4];   // [D,4H]
  const float* Wh  = (const float*)d_in[5];   // [H,4H]
  const float* bb  = (const float*)d_in[6];   // [4H]
  const float* Wf  = (const float*)d_in[7];   // [F,2H]
  const float* bfv = (const float*)d_in[8];   // [2H]

  float* out = (float*)d_out;                 // outputs[B,T,H] ++ h[B,H] ++ c[B,H]

  char* ws = (char*)d_ws;
  size_t off = 0;
  float*  pregate = (float*)(ws + off);  off += (size_t)kBT * kG4 * sizeof(float);
  float*  fadd    = (float*)(ws + off);  off += (size_t)kBT * kH * sizeof(float);
  bf16_t* Whbf    = (bf16_t*)(ws + off); off += (size_t)kD * kG4 * sizeof(bf16_t);
  bf16_t* hbfA    = (bf16_t*)(ws + off); off += (size_t)kB * kH * sizeof(bf16_t);
  bf16_t* hbfB    = (bf16_t*)(ws + off); off += (size_t)kB * kH * sizeof(bf16_t);
  float*  cstate  = (float*)(ws + off);  off += (size_t)kB * kH * sizeof(float);

  // Wh fp32 -> bf16 (8 MB; stays L2-resident across the 256 steps).
  convert_wh_kernel<<<(kD * kG4) / (256 * 4), 256, 0, stream>>>(Wh, Whbf);
  init_state_kernel<<<(kB * kH) / 256, 256, 0, stream>>>(h0, c0, hbfA, cstate);

  // Phase 1: state-independent GEMMs (bulk of the FLOPs, WMMA-bound).
  gemm_xw_kernel<<<dim3(kG4 / 128, kBT / 128), 256, 0, stream>>>(seq, Wx, bb, pregate, kD, kG4);
  gemm_field_kernel<<<dim3(kH / 64, kBT / 128), 256, 0, stream>>>(fld, Wf, bfv, fadd, kD);

  // Phase 2: reverse-time recurrence, one launch per step (stream-ordered).
  for (int s = 0; s < kT; ++s) {
    int t = kT - 1 - s;
    const bf16_t* hs = (s & 1) ? hbfB : hbfA;
    bf16_t*       hd = (s & 1) ? hbfA : hbfB;
    lstm_step_kernel<<<kH / 16, 256, 0, stream>>>(hs, hd, Whbf, pregate, fadd,
                                                  cstate, out, t);
  }

  float* dst_h = out + (size_t)kB * kT * kH;
  float* dst_c = dst_h + (size_t)kB * kH;
  finalize_kernel<<<(kB * kH) / 256, 256, 0, stream>>>(out, cstate, dst_h, dst_c);
}
